// Attention_163208757701
// MI455X (gfx1250) — matrix operand order repaired and checked by reading, not compiled, not run on hardware
//
#include <hip/hip_runtime.h>

// ---------------------------------------------------------------------------
// Problem constants (from reference): b=2, n=2048, d=1024, h=16, hd=64
// ---------------------------------------------------------------------------
#define D_MODEL 1024
#define NHEADS  16
#define HEADD   64
#define SEQ     2048
#define BATCH   2
#define MTOT    (BATCH * SEQ) // 4096 rows

typedef __attribute__((ext_vector_type(16))) __bf16 v16bf;
typedef __attribute__((ext_vector_type(8)))  float  v8f;

union FragBF { v16bf v; uint4 u[2]; };

// ---------------------------------------------------------------------------
// Fragment loaders per CDNA5 ISA 7.12.2 (wave32):
//  A (16x32 bf16): lane m%16 = row; lanes<16 hold K{0..7,16..23} in V0..7,
//                  lanes>=16 hold K{8..15,24..31}.
//  B (32x16 bf16): lane n%16 = column; lanes<16 hold K{0..7,16..23},
//                  lanes>=16 hold K{8..15,24..31} (same as A).
// ---------------------------------------------------------------------------
__device__ __forceinline__ v16bf load_a_frag(const __bf16* tile, int ld, int lane) {
    const int m  = lane & 15;
    const int kb = (lane < 16) ? 0 : 8;
    const __bf16* p = tile + m * ld + kb;
    FragBF f;
    f.u[0] = *reinterpret_cast<const uint4*>(p);        // K kb..kb+7
    f.u[1] = *reinterpret_cast<const uint4*>(p + 16);   // K kb+16..kb+23
    return f.v;
}

__device__ __forceinline__ v16bf load_b_frag(const __bf16* colbase, int cstride, int lane) {
    const int n  = lane & 15;
    const int kb = (lane < 16) ? 0 : 8;
    const __bf16* p = colbase + n * cstride + kb;
    FragBF f;
    f.u[0] = *reinterpret_cast<const uint4*>(p);        // K kb..kb+7
    f.u[1] = *reinterpret_cast<const uint4*>(p + 16);   // K kb+16..kb+23
    return f.v;
}

// All-ones 32x16 B fragment (bf16 1.0 = 0x3F80): P x Ones = row sums of P,
// replicated into every column of the C tile (pre-broadcast row sums).
__device__ __forceinline__ v16bf ones_frag() {
    const unsigned o = 0x3F803F80u;
    FragBF f;
    f.u[0] = uint4{o, o, o, o};
    f.u[1] = uint4{o, o, o, o};
    return f.v;
}

// ---------------------------------------------------------------------------
// gfx1250 async global->LDS copy (16 bytes per lane), tracked by ASYNCcnt.
// LDS byte address = low 32 bits of the flat shared pointer.
// ---------------------------------------------------------------------------
__device__ __forceinline__ void async_copy16(void* lds, const void* gaddr) {
    const unsigned l = (unsigned)(size_t)lds;
    const unsigned long long g = (unsigned long long)(size_t)gaddr;
    asm volatile("global_load_async_to_lds_b128 %0, %1, off"
                 :: "v"(l), "v"(g) : "memory");
}
#define WAIT_ASYNC(n) asm volatile("s_wait_asynccnt " #n ::: "memory")

// ---------------------------------------------------------------------------
// One-time fp32 -> bf16 conversion (x and the three weight matrices).
// ---------------------------------------------------------------------------
__global__ __launch_bounds__(256) void cvt_f32_bf16(const float* __restrict__ s,
                                                    __bf16* __restrict__ d, int n) {
    const int i = (blockIdx.x * 256 + threadIdx.x) * 8;
    if (i + 8 <= n) {
#pragma unroll
        for (int e = 0; e < 8; ++e) d[i + e] = (__bf16)s[i + e];
    }
}

// ---------------------------------------------------------------------------
// GEMM: C[M,Nout] = X[M,K]_bf16 * W[Nout,K]_bf16^T + bias_f32, scaled.
// MODE 0: store bf16 to [B,H,N,HD]      (q, k)
// MODE 1: store bf16 to [B,H,HD,N]      (v transposed, for PV B-fragments)
// MODE 2: store fp32 row-major [M,Nout] (output projection)
// Block: 256 thr = 8 waves (4x2); tile 128(M) x 64(N); K stepped by 32 with
// double-buffered async global->LDS staging.
// ---------------------------------------------------------------------------
template <int MODE>
__global__ __launch_bounds__(256) void gemm_bias_kernel(
    const __bf16* __restrict__ X, const __bf16* __restrict__ W,
    const float* __restrict__ bias, void* __restrict__ out,
    int K, int Nout, float scale)
{
    __shared__ alignas(16) __bf16 sX[2][128 * 32];
    __shared__ alignas(16) __bf16 sW[2][64 * 32];

    const int t    = threadIdx.x;
    const int lane = t & 31;
    const int w    = t >> 5;   // wave 0..7
    const int wm   = w >> 1;   // 0..3  (M sub-tile)
    const int wn   = w & 1;    // 0..1  (N sub-tile)
    const size_t row0g = (size_t)blockIdx.x * 128;
    const size_t col0g = (size_t)blockIdx.y * 64;

    // 128x32 X tile = 512 x 16B chunks; 64x32 W tile = 256 chunks.
    auto stage = [&](int buf, int k0) {
#pragma unroll
        for (int cc = 0; cc < 2; ++cc) {
            const int c = t + cc * 256;            // 0..511
            const int row = c >> 2, part = c & 3;  // 4 chunks per 32-elem row
            async_copy16(&sX[buf][row * 32 + part * 8],
                         X + (row0g + row) * K + k0 + part * 8);
        }
        {
            const int row = t >> 2, part = t & 3;  // 64 rows x 4 chunks
            async_copy16(&sW[buf][row * 32 + part * 8],
                         W + (col0g + row) * K + k0 + part * 8);
        }
    };

    v8f acc[2][2] = {};
    int buf = 0;
    stage(0, 0);

    for (int k0 = 0; k0 < K; k0 += 32) {
        if (k0 + 32 < K) {           // prefetch next tile into other buffer
            stage(buf ^ 1, k0 + 32);
            WAIT_ASYNC(3);           // in-order ASYNCcnt: previous 3 are done
        } else {
            WAIT_ASYNC(0);
        }
        __syncthreads();

        v16bf aF[2], bF[2];
#pragma unroll
        for (int i2 = 0; i2 < 2; ++i2)
            aF[i2] = load_a_frag(sX[buf] + (wm * 32 + i2 * 16) * 32, 32, lane);
#pragma unroll
        for (int j2 = 0; j2 < 2; ++j2)
            bF[j2] = load_b_frag(sW[buf] + (wn * 32 + j2 * 16) * 32, 32, lane);

#pragma unroll
        for (int i2 = 0; i2 < 2; ++i2)
#pragma unroll
            for (int j2 = 0; j2 < 2; ++j2)
                acc[i2][j2] = __builtin_amdgcn_wmma_f32_16x16x32_bf16(
                    false, aF[i2], false, bF[j2], (short)0, acc[i2][j2], false, false);
        __syncthreads();
        buf ^= 1;
    }

    // C-layout: VGPR r <-> row (lane<16 ? r : 8+r), col = lane&15
    const int rbase = (lane < 16) ? 0 : 8;
#pragma unroll
    for (int i2 = 0; i2 < 2; ++i2)
#pragma unroll
        for (int j2 = 0; j2 < 2; ++j2)
#pragma unroll
            for (int r = 0; r < 8; ++r) {
                const int ml = wm * 32 + i2 * 16 + rbase + r;
                const int nl = wn * 32 + j2 * 16 + (lane & 15);
                const size_t gm = row0g + ml;  // global row (b*SEQ+i)
                const size_t gn = col0g + nl;  // output feature
                const float val = (acc[i2][j2][r] + bias[gn]) * scale;
                if (MODE == 2) {
                    ((float*)out)[gm * (size_t)Nout + gn] = val;
                } else {
                    const int bb = (int)(gm >> 11), i = (int)(gm & 2047);
                    const int h  = (int)(gn >> 6),  hd = (int)(gn & 63);
                    __bf16* ob = (__bf16*)out;
                    if (MODE == 0)
                        ob[(((size_t)(bb * NHEADS + h) * SEQ) + i) * HEADD + hd] = (__bf16)val;
                    else // MODE 1: transposed [B,H,HD,N]
                        ob[(((size_t)(bb * NHEADS + h) * HEADD) + hd) * SEQ + i] = (__bf16)val;
                }
            }
}

// ---------------------------------------------------------------------------
// Causal flash attention. q,k: [B,H,N,HD] bf16 (q pre-scaled by hd^-0.5),
// vt: [B,H,HD,N] bf16. Output o: [B*N, D_MODEL] bf16.
// Block = 256 thr = 8 waves = 128 query rows; the block cooperatively
// double-buffers 32-key K/V tiles into LDS via async copies (consumed by all
// 8 waves). Online softmax per wave in the WMMA C-layout; row sums come from
// an extra P x Ones WMMA (pre-broadcast), only row-max uses shuffles.
// Causal masking is applied only on the diagonal key block (scalar branch
// via readfirstlane).
// ---------------------------------------------------------------------------
__global__ __launch_bounds__(256) void attn_kernel(
    const __bf16* __restrict__ q, const __bf16* __restrict__ kk,
    const __bf16* __restrict__ vt, __bf16* __restrict__ o)
{
    __shared__ alignas(16) __bf16 sK[2][32 * 64];  // [key][hd]
    __shared__ alignas(16) __bf16 sV[2][64 * 32];  // [hd][key]
    __shared__ alignas(16) __bf16 sP[8][16 * 32];  // per-wave P staging

    const int t    = threadIdx.x;
    const int lane = t & 31;
    const int w    = t >> 5;                     // wave 0..7
    const int bh   = blockIdx.y;                 // b*NHEADS + h
    const int q0b  = blockIdx.x * 128;           // first query row of block
    const int q0   = q0b + w * 16;               // first query row of wave
    // scalar copy so causal branches compile to s_cbranch, not exec masking
    const int q0s  = __builtin_amdgcn_readfirstlane(q0);

    const __bf16* qp = q  + ((size_t)bh * SEQ + q0) * HEADD;
    const __bf16* kp = kk + (size_t)bh * SEQ * HEADD;
    const __bf16* vp = vt + (size_t)bh * HEADD * SEQ;

    // Q fragments: A-layout, contract dim hd split 0..31 / 32..63
    const v16bf qa0 = load_a_frag(qp,      HEADD, lane);
    const v16bf qa1 = load_a_frag(qp + 32, HEADD, lane);
    const v16bf onesB = ones_frag();

    // Stage 32 keys: K (4KB contiguous) + V^T (64 rows x 64B): 2 issues/thr.
    auto stage = [&](int buf, int j0) {
        const int c = t;                       // 0..255 (16B chunks)
        async_copy16(&sK[buf][c * 8], kp + (size_t)j0 * HEADD + c * 8);
        const int hd = c >> 2, part = c & 3;   // 64 rows x 4 chunks
        async_copy16(&sV[buf][hd * 32 + part * 8],
                     vp + (size_t)hd * SEQ + j0 + part * 8);
    };

    v8f oacc[4] = {};          // O [16 x 64] : 4 hd-tiles of 16
    v8f osum = {};             // running row sums (every column = row sum)
    float mrun[8];
#pragma unroll
    for (int r = 0; r < 8; ++r) mrun[r] = -1e30f;

    const int rbase = (lane < 16) ? 0 : 8;
    const int jend  = q0b + 128;               // block-uniform key horizon

    int buf = 0;
    stage(0, 0);

    for (int j0 = 0; j0 < jend; j0 += 32) {
        if (j0 + 32 < jend) {
            stage(buf ^ 1, j0 + 32);
            WAIT_ASYNC(2);
        } else {
            WAIT_ASYNC(0);
        }
        __syncthreads();

        if (j0 <= q0s + 15) {   // scalar causal guard (EXEC stays all-ones)
            // ---- S = Q * K^T for 32 keys (two 16-key N tiles) ----
            v8f s0 = {}, s1 = {};
            {
                const v16bf b00 = load_b_frag(sK[buf],                   HEADD, lane);
                const v16bf b01 = load_b_frag(sK[buf] + 32,              HEADD, lane);
                const v16bf b10 = load_b_frag(sK[buf] + 16 * HEADD,      HEADD, lane);
                const v16bf b11 = load_b_frag(sK[buf] + 16 * HEADD + 32, HEADD, lane);
                s0 = __builtin_amdgcn_wmma_f32_16x16x32_bf16(false, qa0, false, b00, (short)0, s0, false, false);
                s0 = __builtin_amdgcn_wmma_f32_16x16x32_bf16(false, qa1, false, b01, (short)0, s0, false, false);
                s1 = __builtin_amdgcn_wmma_f32_16x16x32_bf16(false, qa0, false, b10, (short)0, s1, false, false);
                s1 = __builtin_amdgcn_wmma_f32_16x16x32_bf16(false, qa1, false, b11, (short)0, s1, false, false);
            }

            // ---- online softmax (row max via width-16 shuffles) ----
            if (j0 + 31 <= q0s) {
                // fully unmasked block: no compares, no selects
#pragma unroll
                for (int r = 0; r < 8; ++r) {
                    const float x0 = s0[r], x1 = s1[r];
                    float mx = fmaxf(x0, x1);
                    mx = fmaxf(mx, __shfl_xor(mx, 1, 16));
                    mx = fmaxf(mx, __shfl_xor(mx, 2, 16));
                    mx = fmaxf(mx, __shfl_xor(mx, 4, 16));
                    mx = fmaxf(mx, __shfl_xor(mx, 8, 16));
                    const float mnew  = fmaxf(mrun[r], mx);
                    const float alpha = __expf(mrun[r] - mnew);
                    mrun[r] = mnew;
                    const float p0 = __expf(x0 - mnew);
                    const float p1 = __expf(x1 - mnew);
                    osum[r] = osum[r] * alpha;
#pragma unroll
                    for (int tt = 0; tt < 4; ++tt)
                        oacc[tt][r] = oacc[tt][r] * alpha;
                    const int rl = rbase + r;
                    sP[w][rl * 32 + (lane & 15)]      = (__bf16)p0;
                    sP[w][rl * 32 + (lane & 15) + 16] = (__bf16)p1;
                }
            } else {
                // diagonal block: causal mask per element
                const int nb0 = j0 + (lane & 15);
                const int nb1 = nb0 + 16;
#pragma unroll
                for (int r = 0; r < 8; ++r) {
                    const int row = q0 + rbase + r;
                    const float x0 = (nb0 <= row) ? s0[r] : -1e30f;
                    const float x1 = (nb1 <= row) ? s1[r] : -1e30f;
                    float mx = fmaxf(x0, x1);
                    mx = fmaxf(mx, __shfl_xor(mx, 1, 16));
                    mx = fmaxf(mx, __shfl_xor(mx, 2, 16));
                    mx = fmaxf(mx, __shfl_xor(mx, 4, 16));
                    mx = fmaxf(mx, __shfl_xor(mx, 8, 16));
                    const float mnew  = fmaxf(mrun[r], mx);
                    const float alpha = __expf(mrun[r] - mnew);
                    mrun[r] = mnew;
                    const float p0 = __expf(x0 - mnew);   // masked -> 0
                    const float p1 = __expf(x1 - mnew);
                    osum[r] = osum[r] * alpha;
#pragma unroll
                    for (int tt = 0; tt < 4; ++tt)
                        oacc[tt][r] = oacc[tt][r] * alpha;
                    const int rl = rbase + r;
                    sP[w][rl * 32 + (lane & 15)]      = (__bf16)p0;
                    sP[w][rl * 32 + (lane & 15) + 16] = (__bf16)p1;
                }
            }

            // wave-private LDS round trip: drain DS before cross-lane reload
            asm volatile("s_wait_dscnt 0" ::: "memory");
            __builtin_amdgcn_wave_barrier();

            // ---- O += P * V ; row sums += P * Ones (pre-broadcast) ----
            const v16bf pa = load_a_frag(sP[w], 32, lane);
#pragma unroll
            for (int tt = 0; tt < 4; ++tt) {
                const v16bf vb = load_b_frag(sV[buf] + tt * 16 * 32, 32, lane);
                oacc[tt] = __builtin_amdgcn_wmma_f32_16x16x32_bf16(
                    false, pa, false, vb, (short)0, oacc[tt], false, false);
            }
            osum = __builtin_amdgcn_wmma_f32_16x16x32_bf16(
                false, pa, false, onesB, (short)0, osum, false, false);
            asm volatile("s_wait_dscnt 0" ::: "memory");
            __builtin_amdgcn_wave_barrier();
        }

        __syncthreads();
        buf ^= 1;
    }

    // ---- normalize and store [B*N, D_MODEL] bf16 ----
    const int b = bh >> 4, h = bh & 15;
#pragma unroll
    for (int r = 0; r < 8; ++r) {
        const int row = q0 + rbase + r;
        const float rcp = 1.0f / osum[r];
#pragma unroll
        for (int tt = 0; tt < 4; ++tt) {
            const float val = oacc[tt][r] * rcp;
            o[(size_t)(b * SEQ + row) * D_MODEL + h * HEADD + tt * 16 + (lane & 15)] = (__bf16)val;
        }
    }
}

// ---------------------------------------------------------------------------
// Launcher. Workspace (bf16): xbf | Wq | Wkv | Wo | q | k | v^T | attn_out
// = 4M + 1M + 2M + 1M + 4M + 4M + 4M + 4M elems = 48 MB.
// ---------------------------------------------------------------------------
extern "C" void kernel_launch(void* const* d_in, const int* in_sizes, int n_in,
                              void* d_out, int out_size, void* d_ws, size_t ws_size,
                              hipStream_t stream) {
    const float* x   = (const float*)d_in[0];
    const float* Wq  = (const float*)d_in[1];
    const float* bq  = (const float*)d_in[2];
    const float* Wkv = (const float*)d_in[3];
    const float* bkv = (const float*)d_in[4];
    const float* Wo  = (const float*)d_in[5];
    const float* bo  = (const float*)d_in[6];

    const size_t MD = (size_t)MTOT * D_MODEL;      // 4M
    const size_t DD = (size_t)D_MODEL * D_MODEL;   // 1M

    __bf16* xbf  = (__bf16*)d_ws;
    __bf16* wqb  = xbf  + MD;
    __bf16* wkvb = wqb  + DD;
    __bf16* wob  = wkvb + 2 * DD;
    __bf16* qws  = wob  + DD;
    __bf16* kws  = qws  + MD;
    __bf16* vws  = kws  + MD;
    __bf16* ows  = vws  + MD;

    // one-time fp32 -> bf16 conversions (2048 elems per block)
    cvt_f32_bf16<<<MD / 2048,     256, 0, stream>>>(x,   xbf,  (int)MD);
    cvt_f32_bf16<<<DD / 2048,     256, 0, stream>>>(Wq,  wqb,  (int)DD);
    cvt_f32_bf16<<<2 * DD / 2048, 256, 0, stream>>>(Wkv, wkvb, (int)(2 * DD));
    cvt_f32_bf16<<<DD / 2048,     256, 0, stream>>>(Wo,  wob,  (int)DD);

    const dim3 gg(MTOT / 128, D_MODEL / 64);   // 32 x 16 blocks
    const float sm_scale = 0.125f;             // 64^-0.5

    // Q = (x Wq^T + bq) * scale -> [B,H,N,HD]
    gemm_bias_kernel<0><<<gg, 256, 0, stream>>>(xbf, wqb, bq, (void*)qws,
                                                D_MODEL, D_MODEL, sm_scale);
    // K = x Wkv[0:d]^T + bkv[0:d] -> [B,H,N,HD]
    gemm_bias_kernel<0><<<gg, 256, 0, stream>>>(xbf, wkvb, bkv, (void*)kws,
                                                D_MODEL, D_MODEL, 1.0f);
    // V = x Wkv[d:2d]^T + bkv[d:2d] -> transposed [B,H,HD,N]
    gemm_bias_kernel<1><<<gg, 256, 0, stream>>>(xbf, wkvb + DD, bkv + D_MODEL,
                                                (void*)vws, D_MODEL, D_MODEL, 1.0f);

    // causal flash attention -> [B*N, D] bf16
    attn_kernel<<<dim3(SEQ / 128, BATCH * NHEADS), 256, 0, stream>>>(qws, kws, vws, ows);

    // out = attn Wo^T + bo -> fp32
    gemm_bias_kernel<2><<<gg, 256, 0, stream>>>(ows, wob, bo, d_out,
                                                D_MODEL, D_MODEL, 1.0f);
}
